// CTConv2d_75565654606342
// MI455X (gfx1250) — matrix operand order repaired and checked by reading, not compiled, not run on hardware
//
#include <hip/hip_runtime.h>
#include <hip/hip_bf16.h>

// ---------------- types ----------------
typedef __attribute__((ext_vector_type(16))) __bf16 v16bf;
typedef __attribute__((ext_vector_type(8)))  float  v8f;

union AFrag { uint4 u[2]; v16bf v; };
union BFrag { uint4 u[2]; v16bf v; };

#define O_CH 128
#define I_CH 128
#define BATCH 32
#define HW 112
#define KTILES 36            // 9 taps * (128/32)
#define NT 7                 // 112 / 16 output-column tiles
#define CH_HALF 64           // channels staged per LDS pass
#define WPITCH 72            // 64 ch + 8 pad -> 144B = 9*16B (odd multiple of 16B)
#define ROWW 114             // w = -1 .. 112

__device__ __forceinline__ unsigned short f2bf(float f) {
    unsigned int u = __float_as_uint(f);
    u = u + 0x7FFFu + ((u >> 16) & 1u);   // round-to-nearest-even
    return (unsigned short)(u >> 16);
}

// ---------------------------------------------------------------------------
// Kernel 1: synthesize w[o][i][3][3] and scatter into WMMA A-fragment layout.
// A-fragment element map (16-bit A, 16x32, ISA 7.12.2):
//   M = lane&15 ;  K(lane,e) = e + (e>=8 ? 8 : 0) + (lane>=16 ? 8 : 0)
// GEMM K ordering: k = tap*128 + channel ; ktile = tap*4 + isub.
// Apack[((mt*36 + kt)*32 + lane)*16 + e]  (bf16)
// ---------------------------------------------------------------------------
__global__ void ctconv_weight_pack(const float* __restrict__ core,
                                   const float* __restrict__ periphery,
                                   const float* __restrict__ threshold,
                                   const float* __restrict__ scale,
                                   unsigned short* __restrict__ apack) {
    int tid  = blockIdx.x * blockDim.x + threadIdx.x;   // 36*256 = 9216 threads
    int lane = tid & 31;
    int rest = tid >> 5;            // 0..287
    int kt   = rest % KTILES;       // 0..35
    int mt   = rest / KTILES;       // 0..7
    int t    = kt >> 2;             // tap 0..8
    int isub = kt & 3;              // 32-channel group
    int o    = mt * 16 + (lane & 15);

    float thr = threshold[o];
    float sc  = scale[0];
    float pf  = (t == 4) ? 1.0f : periphery[(t < 4) ? t : (t - 1)];

    int lhi = (lane >= 16) ? 8 : 0;
    unsigned long long base =
        ((unsigned long long)(mt * KTILES + kt) * 32ull + (unsigned long long)lane) * 16ull;

#pragma unroll
    for (int e = 0; e < 16; ++e) {
        int kl = e + ((e >= 8) ? 8 : 0) + lhi;      // 0..31
        int i  = isub * 32 + kl;
        float c = core[o * I_CH + i];
        float g = 1.0f / (1.0f + __expf(-sc * (fabsf(c) - thr)));
        float w = (t == 4) ? c : (c * pf * g);
        apack[base + (unsigned)e] = f2bf(w);
    }
}

// ---------------------------------------------------------------------------
// Kernel 2: implicit-GEMM conv. One block per (b, h): 8 waves laid out as
// (mgroup 0..1) x (ngroup 0..3). Each wave: 4 M-tiles x 2 N-tiles, so each
// B fragment fetched from LDS feeds 4 WMMAs (4x less LDS read traffic than
// 1-Mtile-per-wave). ngroup 3 duplicates N-tile 6 (uniform, branch-free WMMA
// path; duplicate results dropped at store time).
// ---------------------------------------------------------------------------
__global__ void __launch_bounds__(256, 2)
ctconv_main(const float* __restrict__ x,
            const unsigned short* __restrict__ apack,
            float* __restrict__ out) {
    __shared__ __align__(16) unsigned short xs[3 * ROWW * WPITCH];  // 49,248 B

    const int b = blockIdx.x / HW;
    const int h = blockIdx.x % HW;

    const int lane  = threadIdx.x & 31;
    const int wv    = threadIdx.x >> 5;      // 0..7
    const int mg    = wv >> 2;               // 0..1 -> mtiles [4mg, 4mg+3]
    const int ng    = wv & 3;                // 0..3
    const int nt0   = (ng < 3) ? (2 * ng) : 6;
    const int nt1   = (ng < 3) ? (2 * ng + 1) : 6;   // duplicate for ng==3
    const int n16   = lane & 15;
    const int lhi16 = (lane >= 16) ? 8 : 0;

    v8f acc[4][2];
#pragma unroll
    for (int j = 0; j < 4; ++j) {
        acc[j][0] = (v8f)(0.0f);
        acc[j][1] = (v8f)(0.0f);
    }

    const uint4* ap4 = (const uint4*)apack;
    const float4* x4 = (const float4*)x;

    for (int half = 0; half < 2; ++half) {
        __syncthreads();   // previous pass's reads done before refill

        // zero the w = -1 and w = 112 pad columns (3 rows x 2 cols x 64 ch)
        for (int idx = threadIdx.x; idx < 3 * 2 * CH_HALF; idx += 256) {
            int ch  = idx & 63;
            int r2  = idx >> 6;            // 0..5
            int row = r2 % 3;
            int w1  = (r2 >= 3) ? (ROWW - 1) : 0;
            xs[(row * ROWW + w1) * WPITCH + ch] = 0;
        }
        // fill interior: 64 ch x 3 rows x 28 float4 (w groups of 4), coalesced
        const int chg = half * CH_HALF;
        for (int u = threadIdx.x; u < CH_HALF * 3 * 28; u += 256) {
            int ch  = u / 84;
            int rem = u % 84;
            int row = rem / 28;
            int w4  = rem % 28;
            int hin = h + row - 1;
            float4 v = make_float4(0.f, 0.f, 0.f, 0.f);
            if (hin >= 0 && hin < HW)
                v = x4[((b * I_CH + chg + ch) * HW + hin) * 28 + w4];
            int base = (row * ROWW + (w4 * 4 + 1)) * WPITCH + ch;
            xs[base              ] = f2bf(v.x);
            xs[base +     WPITCH ] = f2bf(v.y);
            xs[base + 2 * WPITCH ] = f2bf(v.z);
            xs[base + 3 * WPITCH ] = f2bf(v.w);
        }
        __syncthreads();

        // 9 taps x 2 channel-subgroups of this half
        for (int t = 0; t < 9; ++t) {
            const int dh = t / 3;
            const int dw = t % 3;
#pragma unroll
            for (int is2 = 0; is2 < 2; ++is2) {
                const int ktile = t * 4 + half * 2 + is2;

                // A fragments for 4 M-tiles (global, L2-resident, coalesced)
                AFrag a[4];
#pragma unroll
                for (int j = 0; j < 4; ++j) {
                    unsigned long long ai =
                        ((unsigned long long)((mg * 4 + j) * KTILES + ktile) * 32ull +
                         (unsigned long long)lane) * 2ull;
                    a[j].u[0] = ap4[ai];
                    a[j].u[1] = ap4[ai + 1];
                }

                // B: two runs of 8 channels per lane at +0 and +16 (lanes 16-31 at +8 K)
                const int chl = is2 * 32 + lhi16;

                // N-tile 0 of this wave
                {
                    const int w1  = nt0 * 16 + n16 + dw;              // 0..113
                    const int off = (dh * ROWW + w1) * WPITCH + chl;  // 16B aligned
                    BFrag bfr;
                    bfr.u[0] = *(const uint4*)&xs[off];
                    bfr.u[1] = *(const uint4*)&xs[off + 16];
#pragma unroll
                    for (int j = 0; j < 4; ++j)
                        acc[j][0] = __builtin_amdgcn_wmma_f32_16x16x32_bf16(
                            false, a[j].v, false, bfr.v, (short)0, acc[j][0], false, false);
                }
                // N-tile 1 of this wave (duplicate of tile 6 when ng==3)
                {
                    const int w1  = nt1 * 16 + n16 + dw;
                    const int off = (dh * ROWW + w1) * WPITCH + chl;
                    BFrag bfr;
                    bfr.u[0] = *(const uint4*)&xs[off];
                    bfr.u[1] = *(const uint4*)&xs[off + 16];
#pragma unroll
                    for (int j = 0; j < 4; ++j)
                        acc[j][1] = __builtin_amdgcn_wmma_f32_16x16x32_bf16(
                            false, a[j].v, false, bfr.v, (short)0, acc[j][1], false, false);
                }
            }
        }
    }

    // write D: VGPR r, lanes 0-15 -> M=r, N=lane ; lanes 16-31 -> M=r+8, N=lane-16
#pragma unroll
    for (int q = 0; q < 2; ++q) {
        if (q == 1 && ng == 3) continue;          // drop duplicated tile
        const int nt   = q ? nt1 : nt0;
        const int wout = nt * 16 + n16;
#pragma unroll
        for (int j = 0; j < 4; ++j) {
            const int o_base = (mg * 4 + j) * 16 + ((lane >= 16) ? 8 : 0);
#pragma unroll
            for (int r = 0; r < 8; ++r) {
                const int o = o_base + r;
                out[(((unsigned long long)b * O_CH + o) * HW + h) * HW + wout] = acc[j][q][r];
            }
        }
    }
}

// ---------------------------------------------------------------------------
extern "C" void kernel_launch(void* const* d_in, const int* in_sizes, int n_in,
                              void* d_out, int out_size, void* d_ws, size_t ws_size,
                              hipStream_t stream) {
    const float* x         = (const float*)d_in[0];
    const float* core      = (const float*)d_in[1];
    const float* periphery = (const float*)d_in[2];
    const float* threshold = (const float*)d_in[3];
    const float* scale     = (const float*)d_in[4];
    float* out             = (float*)d_out;
    unsigned short* apack  = (unsigned short*)d_ws;   // 294,912 B

    // 8 mtiles * 36 ktiles * 32 lanes = 9216 threads
    ctconv_weight_pack<<<KTILES, 256, 0, stream>>>(core, periphery, threshold, scale, apack);

    ctconv_main<<<BATCH * HW, 256, 0, stream>>>(x, apack, out);
    (void)in_sizes; (void)n_in; (void)out_size; (void)ws_size;
}
